// SAGE_27358941676119
// MI455X (gfx1250) — matrix operand order repaired and checked by reading, not compiled, not run on hardware
//
#include <hip/hip_runtime.h>
#include <hip/hip_bf16.h>
#include <stddef.h>

// ---------------------------------------------------------------------------
// GraphSAGE 2-layer forward for MI455X (gfx1250, wave32, WMMA).
//   Edge aggregation: one wave per edge, float4/lane gathers, native
//   global_atomic_add_f32 via unsafeAtomicAdd (memory-bound part, ~1.3 GB).
//   GEMM: D = [agg_hi|agg_lo|x_hi|x_lo](16x512) x [Wl_hi;Wl_lo;Wr_hi;Wr_lo](512x128)
//   using v_wmma_f32_16x16x32_bf16 with hi/lo bf16 error-split (~f32 accuracy).
// ---------------------------------------------------------------------------

#define D128   128
#define KTOT   512              // 4 * 128 (agg_hi, agg_lo, x_hi, x_lo)
#define LDK    520              // padded halves per LDS row (1040 B, 16B aligned)
#define N1C    200000
#define N2C    49152

typedef __attribute__((ext_vector_type(16))) __bf16 v16bf;
typedef __attribute__((ext_vector_type(8)))  float  v8f;

union Frag32B { uint4 q[2]; v16bf v; };

__device__ __forceinline__ unsigned short f2bf(float f) {
    unsigned u = __float_as_uint(f);
    unsigned r = u + 0x7FFFu + ((u >> 16) & 1u);   // round-to-nearest-even
    return (unsigned short)(r >> 16);
}
__device__ __forceinline__ float bf2f(unsigned short h) {
    return __uint_as_float(((unsigned)h) << 16);
}

// ---------------------------------------------------------------------------
__global__ void sage_zero_f32(float* __restrict__ p, size_t n) {
    size_t i = (size_t)blockIdx.x * blockDim.x + threadIdx.x;
    if (i < n) p[i] = 0.0f;
}

// One wave (32 lanes) per edge; each lane moves 4 floats of the 128-wide row.
// unsafeAtomicAdd -> hardware global_atomic_add_f32 (coarse-grained memory,
// device scope) instead of a CAS loop.
__global__ void sage_edge_agg(const float* __restrict__ feat,
                              const int* __restrict__ src,
                              const int* __restrict__ dst,
                              int nedges,
                              float* __restrict__ sums,
                              float* __restrict__ cnt) {
    int gid  = blockIdx.x * blockDim.x + threadIdx.x;
    int e    = gid >> 5;
    int lane = threadIdx.x & 31;
    if (e >= nedges) return;
    int s = src[e];
    int d = dst[e];
    float4 v = ((const float4*)(feat + (size_t)s * D128))[lane];
    float* o = sums + (size_t)d * D128 + lane * 4;
    unsafeAtomicAdd(o + 0, v.x);
    unsafeAtomicAdd(o + 1, v.y);
    unsafeAtomicAdd(o + 2, v.z);
    unsafeAtomicAdd(o + 3, v.w);
    if (lane == 0)
        unsafeAtomicAdd(cnt + d, 1.0f);
}

// ---------------------------------------------------------------------------
// Persistent GEMM kernel.  256 threads = 8 waves; each wave owns one 16-col
// N-tile of the 128-wide output; workgroup loops over 16-row M-tiles.
// LDS layout (bf16, row stride LDK halves):
//   rows   0..127 : Wt[n][k]   k in [0,512): {Wl_hi, Wl_lo, Wr_hi, Wr_lo}[k%128]
//   rows 128..143 : A tile[m][k]: {agg_hi, agg_lo, x_hi, x_lo}
// ---------------------------------------------------------------------------
__global__ __launch_bounds__(256) void sage_gemm_bf16x2(
        const float* __restrict__ sums, const float* __restrict__ cnt,
        const float* __restrict__ xt,
        const float* __restrict__ Wl,  const float* __restrict__ Wr,
        const float* __restrict__ bias,
        float* __restrict__ out, int ntgt, int do_relu) {
    extern __shared__ unsigned short lds[];
    const int tid  = threadIdx.x;
    const int wave = tid >> 5;
    const int lane = tid & 31;

    // ---- stage split weights into LDS (once per workgroup) ----
    for (int idx = tid; idx < D128 * D128; idx += 256) {
        int k = idx >> 7;          // row of W  (K index)
        int n = idx & 127;         // col of W  (N index)
        float wl = Wl[idx];
        unsigned short hl = f2bf(wl);
        unsigned short ll = f2bf(wl - bf2f(hl));
        float wr = Wr[idx];
        unsigned short hr = f2bf(wr);
        unsigned short lr = f2bf(wr - bf2f(hr));
        unsigned short* row = lds + (size_t)n * LDK;
        row[k]       = hl;
        row[128 + k] = ll;
        row[256 + k] = hr;
        row[384 + k] = lr;
    }
    __syncthreads();

    const int ntiles = ntgt >> 4;            // ntgt is a multiple of 16
    // Fragment base pointers (layouts per CDNA5 ISA 7.12.2):
    //  A 16x32 bf16: lanes 0-15 row=lane  K:{k0..k0+7, k0+16..k0+23}
    //                lanes16-31 row=lane-16 K:{k0+8..k0+15, k0+24..k0+31}
    //  B 32x16 bf16: lanes 0-15 col=lane  K:{k0..k0+7, k0+16..k0+23}
    //                lanes16-31 col=lane-16 K:{k0+8..k0+15, k0+24..k0+31}
    const unsigned short* arow =
        lds + (size_t)(128 + (lane & 15)) * LDK + ((lane >> 4) << 3);
    const unsigned short* brow =
        lds + (size_t)((wave << 4) + (lane & 15)) * LDK + ((lane >> 4) << 3);
    const int   col = (wave << 4) + (lane & 15);
    const float bv  = bias[col];

    for (int mt = blockIdx.x; mt < ntiles; mt += gridDim.x) {
        // ---- build A tile: 16 rows x 128 source-k, split into 512 bf16 ----
        for (int idx = tid; idx < 16 * D128; idx += 256) {
            int m = idx >> 7;
            int k = idx & 127;
            size_t row = (size_t)(mt * 16 + m);
            float c   = cnt[row];
            float inv = 1.0f / fmaxf(c, 1.0f);
            float a   = sums[row * D128 + k] * inv;
            float xv  = xt[row * D128 + k];
            unsigned short ha = f2bf(a);
            unsigned short la = f2bf(a - bf2f(ha));
            unsigned short hx = f2bf(xv);
            unsigned short lx = f2bf(xv - bf2f(hx));
            unsigned short* ar = lds + (size_t)(128 + m) * LDK;
            ar[k]       = ha;
            ar[128 + k] = la;
            ar[256 + k] = hx;
            ar[384 + k] = lx;
        }
        __syncthreads();

        // ---- WMMA accumulate over K = 512 ----
        v8f acc = {};
        #pragma unroll
        for (int k0 = 0; k0 < KTOT; k0 += 32) {
            Frag32B a, b;
            a.q[0] = *(const uint4*)(arow + k0);
            a.q[1] = *(const uint4*)(arow + k0 + 16);
            b.q[0] = *(const uint4*)(brow + k0);
            b.q[1] = *(const uint4*)(brow + k0 + 16);
            acc = __builtin_amdgcn_wmma_f32_16x16x32_bf16(
                      false, a.v, false, b.v, (short)0, acc, false, false);
        }

        // ---- epilogue: bias (+ReLU), store C (VGPR j: rows j / j+8) ----
        int rbase = mt * 16 + ((lane >> 4) << 3);
        #pragma unroll
        for (int j = 0; j < 8; ++j) {
            float v = acc[j] + bv;
            if (do_relu) v = fmaxf(v, 0.0f);
            out[(size_t)(rbase + j) * D128 + col] = v;
        }
        __syncthreads();   // protect A tile before next iteration rewrites it
    }
}

// ---------------------------------------------------------------------------
extern "C" void kernel_launch(void* const* d_in, const int* in_sizes, int n_in,
                              void* d_out, int out_size, void* d_ws, size_t ws_size,
                              hipStream_t stream) {
    const float* x    = (const float*)d_in[0];
    const int*   src0 = (const int*)  d_in[1];
    const int*   dst0 = (const int*)  d_in[2];
    const int*   src1 = (const int*)  d_in[3];
    const int*   dst1 = (const int*)  d_in[4];
    const float* Wl0  = (const float*)d_in[5];
    const float* Wr0  = (const float*)d_in[6];
    const float* b0   = (const float*)d_in[7];
    const float* Wl1  = (const float*)d_in[8];
    const float* Wr1  = (const float*)d_in[9];
    const float* b1   = (const float*)d_in[10];
    const int E0 = in_sizes[1];
    const int E1 = in_sizes[3];

    // workspace: sums0 (also reused as h, in place) | cnt0 | sums1 | cnt1
    float* sums0 = (float*)d_ws;
    float* cnt0  = sums0 + (size_t)N1C * D128;
    float* sums1 = cnt0  + N1C;
    float* cnt1  = sums1 + (size_t)N2C * D128;
    size_t ztot  = (size_t)N1C * (D128 + 1) + (size_t)N2C * (D128 + 1);

    const size_t shmem = (size_t)(D128 + 16) * LDK * sizeof(unsigned short); // 149760 B

    sage_zero_f32<<<(unsigned)((ztot + 255) / 256), 256, 0, stream>>>(sums0, ztot);

    // layer 0 aggregation: one wave per edge
    sage_edge_agg<<<(E0 + 7) / 8, 256, 0, stream>>>(x, src0, dst0, E0, sums0, cnt0);

    // layer 0 GEMM + ReLU, h written in place over sums0
    sage_gemm_bf16x2<<<2048, 256, shmem, stream>>>(
        sums0, cnt0, x, Wl0, Wr0, b0, sums0, N1C, 1);

    // layer 1 aggregation over h (= sums0 buffer)
    sage_edge_agg<<<(E1 + 7) / 8, 256, 0, stream>>>(sums0, src1, dst1, E1, sums1, cnt1);

    // layer 1 GEMM, no ReLU, straight into d_out (h2 flat == three thirds concat)
    sage_gemm_bf16x2<<<1024, 256, shmem, stream>>>(
        sums1, cnt1, sums0, Wl1, Wr1, b1, (float*)d_out, N2C, 0);
}
